// JaddangiAttention_85650237817633
// MI455X (gfx1250) — matrix operand order repaired and checked by reading, not compiled, not run on hardware
//
#include <hip/hip_runtime.h>
#include <hip/hip_bf16.h>
#include <stdint.h>

#define HIDDEN 896
#define NHEADS 14
#define NKV    2
#define NREP   7
#define HDIM   64
#define SEQ    2048
#define NB     2
#define MROWS  (NB * SEQ)   /* 4096 */

typedef __attribute__((ext_vector_type(16))) __bf16       v16bf;
typedef __attribute__((ext_vector_type(8)))  float        v8f;
typedef __attribute__((ext_vector_type(4)))  unsigned int u32x4;

union FragBF {
    uint32_t u[8];
    u32x4    q[2];
    v16bf    v;
};

__device__ __forceinline__ uint32_t bf16_bits(float f) {
    uint32_t x = __float_as_uint(f);
    return (x + 0x7fffu + ((x >> 16) & 1u)) >> 16;   // round-to-nearest-even
}
__device__ __forceinline__ uint32_t pack_bf16(float lo, float hi) {
    return bf16_bits(lo) | (bf16_bits(hi) << 16);
}

// Async global -> LDS copy of one 16-byte chunk (no VGPR data movement).
// Flat pointers into the LDS aperture carry the LDS offset in their low 32 bits
// (CDNA5 ISA 10.2).
__device__ __forceinline__ void async_copy_b128(const void* g, void* s) {
    uint32_t lds = (uint32_t)(uintptr_t)s;
    uint64_t ga  = (uint64_t)(uintptr_t)g;
    asm volatile("global_load_async_to_lds_b128 %0, %1, off"
                 :: "v"(lds), "v"(ga) : "memory");
}
__device__ __forceinline__ void wait_asynccnt0() {
    asm volatile("s_wait_asynccnt 0" ::: "memory");
}

// A-matrix fragment: 16x32 bf16 tile, source row-major [m][k], ld elements.
// Lane m = lane&15, half = lane>>4.  V0-3: K = half*8 + 0..7 ; V4-7: K = 16 + half*8 + 0..7.
__device__ __forceinline__ FragBF load_a_bf16(const uint16_t* base, int ld, int kbase, int lane) {
    int row = lane & 15;
    int k0  = kbase + ((lane >> 4) << 3);
    FragBF f;
    f.q[0] = *reinterpret_cast<const u32x4*>(base + row * ld + k0);
    f.q[1] = *reinterpret_cast<const u32x4*>(base + row * ld + k0 + 16);
    return f;
}

// B-matrix fragment: 32x16 bf16, source stored n-major [n][k] row-major.
// Lane n = lane&15, half = lane>>4.  V0-3: K = half*8 + 0..7 ; V4-7: K = 16 + half*8 + 0..7.
__device__ __forceinline__ FragBF load_b_bf16(const uint16_t* base, int ld, int nbase, int kbase, int lane) {
    int row = nbase + (lane & 15);
    int k0  = kbase + ((lane >> 4) << 3);
    FragBF f;
    f.q[0] = *reinterpret_cast<const u32x4*>(base + row * ld + k0);
    f.q[1] = *reinterpret_cast<const u32x4*>(base + row * ld + k0 + 16);
    return f;
}

__device__ __forceinline__ v8f wmma_bf16(const FragBF& a, const FragBF& b, v8f c) {
    return __builtin_amdgcn_wmma_f32_16x16x32_bf16(
        /*neg_a=*/false, a.v, /*neg_b=*/false, b.v,
        /*c_mod=*/(short)0, c, /*reuse_a=*/false, /*reuse_b=*/false);
}

// -------------------------------------------------------------------------
// Kernel 0: one-time f32 -> bf16 conversion (pairs -> packed dwords).
// -------------------------------------------------------------------------
__global__ __launch_bounds__(256) void cvt_bf16_kernel(
    const float* __restrict__ src, uint16_t* __restrict__ dst, int n2)
{
    int i = blockIdx.x * blockDim.x + threadIdx.x;
    if (i < n2)
        reinterpret_cast<uint32_t*>(dst)[i] = pack_bf16(src[2 * i], src[2 * i + 1]);
}

// -------------------------------------------------------------------------
// Kernel 1: fused QKV projection + RoPE (all-bf16 operands).
// grid = (MROWS/16, 2), block = 288 (9 waves).  n-tile id = y*9 + wave:
//   0..13 -> Q head, 14..15 -> K kv-head, 16..17 -> V kv-head.
// Outputs: Q bf16 [B][NH][S][64], K bf16 [B][NKV][S][64], Vt bf16 [B][NKV][64][S].
// -------------------------------------------------------------------------
__global__ __launch_bounds__(288) void qkv_rope_kernel(
    const uint16_t* __restrict__ hsb, const int* __restrict__ pos_ids,
    const uint16_t* __restrict__ Wqb, const float* __restrict__ bq,
    const uint16_t* __restrict__ Wkb, const float* __restrict__ bk,
    const uint16_t* __restrict__ Wvb, const float* __restrict__ bv,
    uint16_t* __restrict__ Qb, uint16_t* __restrict__ Kb, uint16_t* __restrict__ Vt)
{
    __shared__ __align__(16) uint16_t sA[16 * HIDDEN];
    const int m0 = blockIdx.x * 16;

    {   // async-DMA stage of the 16x896 bf16 row block into LDS (1792 x 16B)
        const uint16_t* src = hsb + (size_t)m0 * HIDDEN;
        for (int i = threadIdx.x; i < (16 * HIDDEN) / 8; i += blockDim.x)
            async_copy_b128(src + i * 8, &sA[i * 8]);
        wait_asynccnt0();
    }
    __syncthreads();

    const int wid = threadIdx.x >> 5, lane = threadIdx.x & 31;
    const int ntile = blockIdx.y * 9 + wid;
    if (ntile >= 18) return;

    int kind, head, nb0;
    const uint16_t* W;
    const float* bias;
    if (ntile < 14)      { kind = 0; head = ntile;      nb0 = ntile * 64;        W = Wqb; bias = bq; }
    else if (ntile < 16) { kind = 1; head = ntile - 14; nb0 = (ntile - 14) * 64; W = Wkb; bias = bk; }
    else                 { kind = 2; head = ntile - 16; nb0 = (ntile - 16) * 64; W = Wvb; bias = bv; }

    v8f acc[4] = {};
    for (int kb = 0; kb < HIDDEN; kb += 32) {
        FragBF a = load_a_bf16(sA, HIDDEN, kb, lane);
#pragma unroll
        for (int j = 0; j < 4; ++j) {
            FragBF bf = load_b_bf16(W, HIDDEN, nb0 + j * 16, kb, lane);
            acc[j] = wmma_bf16(a, bf, acc[j]);
        }
    }

    const int nlane = lane & 15, halfid = lane >> 4;
#pragma unroll
    for (int j = 0; j < 4; ++j) {
        float bv_ = bias[nb0 + j * 16 + nlane];
#pragma unroll
        for (int r = 0; r < 8; ++r) acc[j][r] += bv_;
    }

    const int b    = m0 >> 11;          // 2048 rows per batch
    const int sblk = m0 & (SEQ - 1);

    if (kind < 2) {
        // RoPE: dim pairs (d, d+32) live in (acc[j], acc[j+2]) at the same lane.
        float invf[2];
#pragma unroll
        for (int j = 0; j < 2; ++j) {
            float d = (float)(j * 16 + nlane);            // d in [0,32)
            invf[j] = __expf(d * -0.43173470f);           // theta^(-2d/64), ln(1e6)/32
        }
#pragma unroll
        for (int r = 0; r < 8; ++r) {
            int srow  = sblk + r + 8 * halfid;
            float pos = (float)pos_ids[b * SEQ + srow];
#pragma unroll
            for (int j = 0; j < 2; ++j) {
                float f  = pos * invf[j];
                float cv = __cosf(f), sv = __sinf(f);
                float lo = acc[j][r], hi = acc[j + 2][r];
                acc[j][r]     = lo * cv - hi * sv;
                acc[j + 2][r] = hi * cv + lo * sv;
            }
        }
        uint16_t* dst = (kind == 0) ? Qb : Kb;
        size_t base = (size_t)(b * (kind == 0 ? NHEADS : NKV) + head) * SEQ * HDIM;
#pragma unroll
        for (int r = 0; r < 8; ++r) {
            size_t srow = (size_t)(sblk + r + 8 * halfid);
#pragma unroll
            for (int j = 0; j < 4; ++j)
                dst[base + srow * HDIM + j * 16 + nlane] = (uint16_t)bf16_bits(acc[j][r]);
        }
    } else {
        // V stored transposed: Vt[b][kv][d][s]
        size_t base = (size_t)(b * NKV + head) * HDIM * SEQ;
#pragma unroll
        for (int r = 0; r < 8; ++r) {
            int srow = sblk + r + 8 * halfid;
#pragma unroll
            for (int j = 0; j < 4; ++j)
                Vt[base + (size_t)(j * 16 + nlane) * SEQ + srow] = (uint16_t)bf16_bits(acc[j][r]);
        }
    }
}

// -------------------------------------------------------------------------
// Kernel 2: causal flash attention.  One wave per 16-query tile, 4 waves/block.
// grid = B*NH*(S/16)/4 = 896 blocks, block = 128 threads.
// -------------------------------------------------------------------------
__global__ __launch_bounds__(128) void attn_kernel(
    const uint16_t* __restrict__ Qb, const uint16_t* __restrict__ Kb,
    const uint16_t* __restrict__ Vt, uint16_t* __restrict__ Ab)
{
    __shared__ __align__(16) uint16_t sP[4][16 * 64];   // per-wave P relayout tile

    const int wid  = threadIdx.x >> 5, lane = threadIdx.x & 31;
    const int wg   = blockIdx.x * 4 + wid;
    const int qt   = wg & 127;            // 128 q-tiles per (b,h)
    const int t    = wg >> 7;
    const int h    = t % NHEADS, b = t / NHEADS;
    const int kv   = h / NREP;
    const int q0   = qt * 16;
    const int nlane = lane & 15, halfid = lane >> 4;

    const uint16_t* Qp = Qb + ((size_t)(b * NHEADS + h) * SEQ + q0) * HDIM;
    const uint16_t* Kp = Kb + (size_t)(b * NKV + kv) * SEQ * HDIM;
    const uint16_t* Vp = Vt + (size_t)(b * NKV + kv) * HDIM * SEQ;

    FragBF qa0 = load_a_bf16(Qp, HDIM, 0, lane);
    FragBF qa1 = load_a_bf16(Qp, HDIM, 32, lane);

    v8f o[4] = {};
    float mrow[8], lrow[8];
#pragma unroll
    for (int r = 0; r < 8; ++r) { mrow[r] = -3.0e38f; lrow[r] = 0.0f; }

    uint16_t* sPw = &sP[wid][0];

    for (int kb = 0; kb <= q0 + 15; kb += 64) {
        // ---- S = Q * K^T (scaled) ----
        v8f sc[4] = {};
#pragma unroll
        for (int jn = 0; jn < 4; ++jn) {
            FragBF k0f = load_b_bf16(Kp, HDIM, kb + jn * 16, 0, lane);
            FragBF k1f = load_b_bf16(Kp, HDIM, kb + jn * 16, 32, lane);
            sc[jn] = wmma_bf16(qa0, k0f, sc[jn]);
            sc[jn] = wmma_bf16(qa1, k1f, sc[jn]);
        }
        const bool needmask = (kb + 63 > q0);
#pragma unroll
        for (int jn = 0; jn < 4; ++jn) {
#pragma unroll
            for (int r = 0; r < 8; ++r) {
                float v = sc[jn][r] * 0.125f;           // 1/sqrt(64)
                if (needmask) {
                    int col  = kb + jn * 16 + nlane;
                    int rowg = q0 + r + 8 * halfid;
                    if (col > rowg) v = -3.0e38f;
                }
                sc[jn][r] = v;
            }
        }
        // ---- online softmax (row = r + 8*half spread across a 16-lane group) ----
#pragma unroll
        for (int r = 0; r < 8; ++r) {
            float t0 = fmaxf(fmaxf(sc[0][r], sc[1][r]), fmaxf(sc[2][r], sc[3][r]));
#pragma unroll
            for (int mk = 8; mk >= 1; mk >>= 1)
                t0 = fmaxf(t0, __shfl_xor(t0, mk, 16));
            float mnew = fmaxf(mrow[r], t0);
            float fac  = __expf(mrow[r] - mnew);
            mrow[r] = mnew;
#pragma unroll
            for (int j = 0; j < 4; ++j) o[j][r] *= fac;
            float rs = 0.0f;
#pragma unroll
            for (int j = 0; j < 4; ++j) {
                float p = __expf(sc[j][r] - mnew);
                sc[j][r] = p;
                rs += p;
            }
#pragma unroll
            for (int mk = 8; mk >= 1; mk >>= 1)
                rs += __shfl_xor(rs, mk, 16);
            lrow[r] = lrow[r] * fac + rs;
        }
        // ---- re-layout P: C-fragment -> A-fragment via per-wave LDS tile ----
#pragma unroll
        for (int j = 0; j < 4; ++j)
#pragma unroll
            for (int r = 0; r < 8; ++r)
                sPw[(r + 8 * halfid) * 64 + j * 16 + nlane] = (uint16_t)bf16_bits(sc[j][r]);
        asm volatile("s_wait_dscnt 0" ::: "memory");
        FragBF pa0 = load_a_bf16(sPw, 64, 0, lane);
        FragBF pa1 = load_a_bf16(sPw, 64, 32, lane);
        // ---- O += P * V  (B-fragments from transposed V: n = d, k = key) ----
#pragma unroll
        for (int jn = 0; jn < 4; ++jn) {
            FragBF v0f = load_b_bf16(Vp, SEQ, jn * 16, kb, lane);
            FragBF v1f = load_b_bf16(Vp, SEQ, jn * 16, kb + 32, lane);
            o[jn] = wmma_bf16(pa0, v0f, o[jn]);
            o[jn] = wmma_bf16(pa1, v1f, o[jn]);
        }
    }

    // ---- finalize: O /= l, store to attn buffer [B][S][NH*64] bf16 ----
#pragma unroll
    for (int r = 0; r < 8; ++r) {
        float inv  = 1.0f / lrow[r];
        int   srow = q0 + r + 8 * halfid;
        size_t base = ((size_t)b * SEQ + srow) * HIDDEN + h * HDIM;
#pragma unroll
        for (int j = 0; j < 4; ++j)
            Ab[base + j * 16 + nlane] = (uint16_t)bf16_bits(o[j][r] * inv);
    }
}

// -------------------------------------------------------------------------
// Kernel 3: output projection  out = attn @ Wo^T   (M=4096, N=896, K=896)
// grid = (MROWS/16, 2), block = 224 (7 waves); n-tile = y*7 + wave (0..13).
// -------------------------------------------------------------------------
__global__ __launch_bounds__(224) void oproj_kernel(
    const uint16_t* __restrict__ Ab, const uint16_t* __restrict__ Wob,
    float* __restrict__ out)
{
    __shared__ __align__(16) uint16_t sA[16 * HIDDEN];
    const int m0 = blockIdx.x * 16;
    {   // async-DMA stage of the 16x896 bf16 row block into LDS
        const uint16_t* src = Ab + (size_t)m0 * HIDDEN;
        for (int i = threadIdx.x; i < (16 * HIDDEN) / 8; i += blockDim.x)
            async_copy_b128(src + i * 8, &sA[i * 8]);
        wait_asynccnt0();
    }
    __syncthreads();

    const int wid = threadIdx.x >> 5, lane = threadIdx.x & 31;
    const int ntile = blockIdx.y * 7 + wid;     // 0..13
    const int nb0 = ntile * 64;

    v8f acc[4] = {};
    for (int kb = 0; kb < HIDDEN; kb += 32) {
        FragBF a = load_a_bf16(sA, HIDDEN, kb, lane);
#pragma unroll
        for (int j = 0; j < 4; ++j) {
            FragBF bf = load_b_bf16(Wob, HIDDEN, nb0 + j * 16, kb, lane);
            acc[j] = wmma_bf16(a, bf, acc[j]);
        }
    }

    const int nlane = lane & 15, halfid = lane >> 4;
#pragma unroll
    for (int r = 0; r < 8; ++r) {
        size_t row = (size_t)(m0 + r + 8 * halfid);
#pragma unroll
        for (int j = 0; j < 4; ++j)
            out[row * HIDDEN + nb0 + j * 16 + nlane] = acc[j][r];
    }
}

// -------------------------------------------------------------------------
extern "C" void kernel_launch(void* const* d_in, const int* in_sizes, int n_in,
                              void* d_out, int out_size, void* d_ws, size_t ws_size,
                              hipStream_t stream)
{
    const float* hs  = (const float*)d_in[0];
    const int*   pos = (const int*)  d_in[1];
    const float* Wq  = (const float*)d_in[2];
    const float* bq  = (const float*)d_in[3];
    const float* Wk  = (const float*)d_in[4];
    const float* bk  = (const float*)d_in[5];
    const float* Wv  = (const float*)d_in[6];
    const float* bv  = (const float*)d_in[7];
    const float* Wo  = (const float*)d_in[8];
    float* out = (float*)d_out;

    // Workspace layout (bytes, all 16B-aligned):
    //   hsb 7340032 | Wqb 1605632 | Wkb 229376 | Wvb 229376 | Wob 1605632 |
    //   Qb 7340032 | Kb 1048576 | Vt 1048576 | Ab 7340032          (~27.8 MB)
    uint8_t* ws = (uint8_t*)d_ws;
    uint16_t* hsb = (uint16_t*)(ws);
    uint16_t* Wqb = (uint16_t*)(ws + (size_t)7340032);
    uint16_t* Wkb = (uint16_t*)(ws + (size_t)8945664);
    uint16_t* Wvb = (uint16_t*)(ws + (size_t)9175040);
    uint16_t* Wob = (uint16_t*)(ws + (size_t)9404416);
    uint16_t* Qb  = (uint16_t*)(ws + (size_t)11010048);
    uint16_t* Kb  = (uint16_t*)(ws + (size_t)18350080);
    uint16_t* Vt  = (uint16_t*)(ws + (size_t)19398656);
    uint16_t* Ab  = (uint16_t*)(ws + (size_t)20447232);

    // One-time f32 -> bf16 conversions.
    {
        const int nhs = MROWS * HIDDEN / 2;          // 1835008
        const int nwq = HIDDEN * HIDDEN / 2;         // 401408
        const int nwk = (NKV * HDIM) * HIDDEN / 2;   // 57344
        cvt_bf16_kernel<<<dim3((nhs + 255) / 256), dim3(256), 0, stream>>>(hs, hsb, nhs);
        cvt_bf16_kernel<<<dim3((nwq + 255) / 256), dim3(256), 0, stream>>>(Wq, Wqb, nwq);
        cvt_bf16_kernel<<<dim3((nwk + 255) / 256), dim3(256), 0, stream>>>(Wk, Wkb, nwk);
        cvt_bf16_kernel<<<dim3((nwk + 255) / 256), dim3(256), 0, stream>>>(Wv, Wvb, nwk);
        cvt_bf16_kernel<<<dim3((nwq + 255) / 256), dim3(256), 0, stream>>>(Wo, Wob, nwq);
    }

    qkv_rope_kernel<<<dim3(MROWS / 16, 2), dim3(288), 0, stream>>>(
        hsb, pos, Wqb, bq, Wkb, bk, Wvb, bv, Qb, Kb, Vt);

    attn_kernel<<<dim3((NB * NHEADS * (SEQ / 16)) / 4), dim3(128), 0, stream>>>(
        Qb, Kb, Vt, Ab);

    oproj_kernel<<<dim3(MROWS / 16, 2), dim3(224), 0, stream>>>(Ab, Wob, out);
}
